// UNet_36996848288217
// MI455X (gfx1250) — matrix operand order repaired and checked by reading, not compiled, not run on hardware
//
#include <hip/hip_runtime.h>
#include <hip/hip_bf16.h>

#define BD    2
#define NPTS  150000
#define CCH   64
#define DD    128
#define D3    (DD * DD * DD)
#define TAPS  27
#define EPSV  1e-4f
#define NTILES (NPTS / 16)          // 9375 tiles of 16 points
#define WPL   (TAPS * 4096)         // wpack halves per layer: 27*2*4*32*16
#define CONV_WAVES 16               // waves per conv workgroup (512 threads)

typedef __attribute__((ext_vector_type(16))) _Float16 v16h;
typedef __attribute__((ext_vector_type(8)))  float    v8f;

union FragA { v16h h; uint4 u[2]; };

// ---------------- setup kernels ----------------

__global__ void fill_i32(int* __restrict__ p, int v, int n) {
    int i = blockIdx.x * blockDim.x + threadIdx.x;
    if (i < n) p[i] = v;
}

__global__ void zero_f32(float* __restrict__ p, int n) {
    int i = blockIdx.x * blockDim.x + threadIdx.x;
    if (i < n) p[i] = 0.f;
}

__global__ void scatter_lut(const int* __restrict__ pos, int* __restrict__ lut) {
    int i = blockIdx.x * blockDim.x + threadIdx.x;   // b*N + n
    if (i >= BD * NPTS) return;
    int b = i / NPTS, n = i - b * NPTS;
    const int* pp = pos + (size_t)i * 3;
    int code = (pp[0] * DD + pp[1]) * DD + pp[2];
    lut[b * D3 + code] = n;
}

__global__ void build_nbr(const int* __restrict__ pos, const int* __restrict__ lut,
                          int* __restrict__ nbr) {
    int i = blockIdx.x * blockDim.x + threadIdx.x;   // b*N + n
    if (i >= BD * NPTS) return;
    int b = i / NPTS;
    const int* pp = pos + (size_t)i * 3;
    int x = pp[0], y = pp[1], z = pp[2];
    int* out = nbr + (size_t)i * TAPS;
    for (int t = 0; t < TAPS; ++t) {
        int dx = t / 9 - 1, dy = (t / 3) % 3 - 1, dz = t % 3 - 1;
        int nx = x + dx, ny = y + dy, nz = z + dz;
        int v = -1;
        if ((unsigned)nx < DD && (unsigned)ny < DD && (unsigned)nz < DD)
            v = lut[b * D3 + (nx * DD + ny) * DD + nz];
        out[t] = v;
    }
}

// Pack weights [4][27][Cin=64][Cout=64] f32 into per-lane WMMA B-fragment
// layout: [layer][t][kchunk c(2)][ntile j(4)][lane(32)][16 halves]
// B 32x16 f16 layout: lane L -> n = j*16 + (L&15), k = c*32 + (L>>4)*8 + (jj&7) + (jj>>3)*16
__global__ void pack_w(const float* __restrict__ W, _Float16* __restrict__ wpack) {
    int i = blockIdx.x * blockDim.x + threadIdx.x;
    const int total = 4 * TAPS * 4096;
    if (i >= total) return;
    int jj   = i & 15;
    int lane = (i >> 4) & 31;
    int j    = (i >> 9) & 3;
    int c    = (i >> 11) & 1;
    int lt   = i >> 12;                       // layer*27 + t
    int n = j * 16 + (lane & 15);
    int k = c * 32 + (lane >> 4) * 8 + (jj & 7) + (jj >> 3) * 16;
    wpack[i] = (_Float16)W[((size_t)lt * CCH + k) * CCH + n];
}

// ---------------- batchnorm ----------------

__global__ void stats_kernel(const float* __restrict__ x, float* __restrict__ stat) {
    __shared__ float ss[256], sq[256];
    const int total = BD * NPTS * CCH;
    int tid = blockIdx.x * 256 + threadIdx.x;
    float s = 0.f, q = 0.f;
    for (int i = tid; i < total; i += gridDim.x * 256) {  // stride % 64 == 0 -> channel fixed
        float v = x[i];
        s += v; q += v * v;
    }
    ss[threadIdx.x] = s; sq[threadIdx.x] = q;
    __syncthreads();
    if (threadIdx.x < CCH) {
        int c = threadIdx.x;
        float S = ss[c] + ss[c + 64] + ss[c + 128] + ss[c + 192];
        float Q = sq[c] + sq[c + 64] + sq[c + 128] + sq[c + 192];
        atomicAdd(&stat[c], S);
        atomicAdd(&stat[CCH + c], Q);
    }
}

__global__ void finalize_bn(const float* __restrict__ stat,
                            const float* __restrict__ gamma, const float* __restrict__ beta,
                            float* __restrict__ sb) {
    int c = threadIdx.x;
    if (c < CCH) {
        const float inv = 1.f / (float)(BD * NPTS);
        float mu  = stat[c] * inv;
        float var = stat[CCH + c] * inv - mu * mu;
        float sc  = gamma[c] * rsqrtf(var + EPSV);
        sb[c]       = sc;
        sb[CCH + c] = beta[c] - mu * sc;
    }
}

__global__ void bn_relu_f16(const float* __restrict__ x, const float* __restrict__ sb,
                            _Float16* __restrict__ out) {
    int i = blockIdx.x * blockDim.x + threadIdx.x;
    const int total = BD * NPTS * CCH;
    if (i >= total) return;
    int c = i & (CCH - 1);
    float v = x[i] * sb[c] + sb[CCH + c];
    out[i] = (_Float16)(v > 0.f ? v : 0.f);
}

__global__ void add_f32(const float* __restrict__ a, const float* __restrict__ b,
                        float* __restrict__ o) {
    int i = blockIdx.x * blockDim.x + threadIdx.x;
    const int total = BD * NPTS * CCH;
    if (i >= total) return;
    o[i] = a[i] + b[i];
}

// ---------------- WMMA submanifold conv ----------------
// 16 waves per workgroup; full per-layer weight pack (216 KB) staged in LDS
// (CDNA5: up to 320 KB LDS per workgroup). Each wave computes one 16-point
// tile -> 16x64 output slab via 4 f32 accumulators; K = 27 taps x 64 ch.

__global__ __launch_bounds__(CONV_WAVES * 32) void conv_wmma(
    const _Float16* __restrict__ h,      // [B,N,64] f16
    const _Float16* __restrict__ wpack,  // [27][2][4][32][16] f16 for this layer
    const int*      __restrict__ nbr,    // [B,N,27]
    float*          __restrict__ y)      // [B,N,64] f32
{
    __shared__ _Float16 wlds[WPL];       // 221,184 bytes

    // cooperative stage: global b128 -> LDS b128
    {
        const uint4* src = (const uint4*)wpack;
        uint4*       dst = (uint4*)wlds;
        for (int i = threadIdx.x; i < WPL / 8; i += CONV_WAVES * 32)
            dst[i] = src[i];
    }
    __syncthreads();

    int wave = threadIdx.x >> 5;
    int lane = threadIdx.x & 31;
    int wid  = blockIdx.x * CONV_WAVES + wave;   // global tile id
    if (wid >= BD * NTILES) return;              // wave-uniform tail guard

    int b    = wid / NTILES;
    int tile = wid - b * NTILES;
    int mrow = lane & 15;
    int hi   = lane >> 4;
    int p    = tile * 16 + mrow;

    const int*      nrow = nbr + ((size_t)b * NPTS + p) * TAPS;
    const _Float16* hb   = h + (size_t)b * NPTS * CCH;

    v8f acc0 = {}, acc1 = {}, acc2 = {}, acc3 = {};
    const int ka = hi * 8;               // A-fragment per-lane K offset

    for (int t = 0; t < TAPS; ++t) {
        int nidx = nrow[t];
        FragA a0, a1;
        if (nidx >= 0) {
            const uint4* row = (const uint4*)(hb + (size_t)nidx * CCH); // 64 halves = 8x b128
            a0.u[0] = row[(ka)      >> 3];   // K [ka,      ka+8)
            a0.u[1] = row[(16 + ka) >> 3];   // K [16+ka,   16+ka+8)
            a1.u[0] = row[(32 + ka) >> 3];   // K [32+ka, ...)
            a1.u[1] = row[(48 + ka) >> 3];
        } else {
            uint4 z = {0u, 0u, 0u, 0u};
            a0.u[0] = z; a0.u[1] = z; a1.u[0] = z; a1.u[1] = z;
        }
        const v16h* wl = (const v16h*)(wlds + t * 4096);  // LDS B fragments
        // kchunk 0 (K=0..31)
        acc0 = __builtin_amdgcn_wmma_f32_16x16x32_f16(false, a0.h, false, wl[(0 * 4 + 0) * 32 + lane], (short)0, acc0, false, false);
        acc1 = __builtin_amdgcn_wmma_f32_16x16x32_f16(false, a0.h, false, wl[(0 * 4 + 1) * 32 + lane], (short)0, acc1, false, false);
        acc2 = __builtin_amdgcn_wmma_f32_16x16x32_f16(false, a0.h, false, wl[(0 * 4 + 2) * 32 + lane], (short)0, acc2, false, false);
        acc3 = __builtin_amdgcn_wmma_f32_16x16x32_f16(false, a0.h, false, wl[(0 * 4 + 3) * 32 + lane], (short)0, acc3, false, false);
        // kchunk 1 (K=32..63)
        acc0 = __builtin_amdgcn_wmma_f32_16x16x32_f16(false, a1.h, false, wl[(1 * 4 + 0) * 32 + lane], (short)0, acc0, false, false);
        acc1 = __builtin_amdgcn_wmma_f32_16x16x32_f16(false, a1.h, false, wl[(1 * 4 + 1) * 32 + lane], (short)0, acc1, false, false);
        acc2 = __builtin_amdgcn_wmma_f32_16x16x32_f16(false, a1.h, false, wl[(1 * 4 + 2) * 32 + lane], (short)0, acc2, false, false);
        acc3 = __builtin_amdgcn_wmma_f32_16x16x32_f16(false, a1.h, false, wl[(1 * 4 + 3) * 32 + lane], (short)0, acc3, false, false);
    }

    // C/D 16x16 f32 layout: vgpr r -> m = r + hi*8 ; n = lane&15
    float* yb = y + (size_t)b * NPTS * CCH;
    int ncol = lane & 15;
#pragma unroll
    for (int r = 0; r < 8; ++r) {
        int po = tile * 16 + r + hi * 8;
        float* yr = yb + (size_t)po * CCH + ncol;
        yr[0]  = acc0[r];
        yr[16] = acc1[r];
        yr[32] = acc2[r];
        yr[48] = acc3[r];
    }
}

// ---------------- host orchestration ----------------

extern "C" void kernel_launch(void* const* d_in, const int* in_sizes, int n_in,
                              void* d_out, int out_size, void* d_ws, size_t ws_size,
                              hipStream_t stream) {
    const float* feats  = (const float*)d_in[0];   // [B,N,64]
    const float* Ws     = (const float*)d_in[1];   // [4,27,64,64]
    const float* gammas = (const float*)d_in[2];   // [4,64]
    const float* betas  = (const float*)d_in[3];   // [4,64]
    const int*   pos    = (const int*)d_in[4];     // [B,N,3]

    char*  ws  = (char*)d_ws;
    size_t off = 0;
    auto alloc = [&](size_t bytes) -> void* {
        void* p = ws + off;
        off = (off + bytes + 255) & ~(size_t)255;
        return p;
    };
    int*       lut   = (int*)alloc((size_t)BD * D3 * 4);                // 16 MB
    int*       nbr   = (int*)alloc((size_t)BD * NPTS * TAPS * 4);       // 32.4 MB
    _Float16*  h16   = (_Float16*)alloc((size_t)BD * NPTS * CCH * 2);   // 38.4 MB
    float*     ybuf  = (float*)alloc((size_t)BD * NPTS * CCH * 4);      // 76.8 MB
    _Float16*  wpack = (_Float16*)alloc((size_t)4 * WPL * 2);           // 0.9 MB
    float*     stat  = (float*)alloc(2 * CCH * 4);                      // sum | sumsq
    float*     sb    = (float*)alloc(2 * CCH * 4);                      // scale | bias
    float*     xout  = (float*)d_out;                                   // live x buffer

    const int totalF = BD * NPTS * CCH;
    const int ewGrid = (totalF + 255) / 256;
    const int convGrid = (BD * NTILES + CONV_WAVES - 1) / CONV_WAVES;   // 1172

    // setup
    fill_i32<<<(BD * D3 + 255) / 256, 256, 0, stream>>>(lut, -1, BD * D3);
    scatter_lut<<<(BD * NPTS + 255) / 256, 256, 0, stream>>>(pos, lut);
    build_nbr<<<(BD * NPTS + 255) / 256, 256, 0, stream>>>(pos, lut, nbr);
    pack_w<<<(4 * WPL + 255) / 256, 256, 0, stream>>>(Ws, wpack);

    auto bn_stage = [&](const float* xin, int layer) {
        zero_f32<<<1, 128, 0, stream>>>(stat, 2 * CCH);
        stats_kernel<<<1024, 256, 0, stream>>>(xin, stat);
        finalize_bn<<<1, 64, 0, stream>>>(stat, gammas + layer * CCH, betas + layer * CCH, sb);
        bn_relu_f16<<<ewGrid, 256, 0, stream>>>(xin, sb, h16);
    };
    auto conv_stage = [&](int layer) {
        conv_wmma<<<convGrid, CONV_WAVES * 32, 0, stream>>>(
            h16, wpack + (size_t)layer * WPL, nbr, ybuf);
    };

    // residual block 1 (x = feats)
    bn_stage(feats, 0);
    conv_stage(0);                 // ybuf = conv1(h)
    bn_stage(ybuf, 1);
    conv_stage(1);                 // ybuf = conv2(h)  (overwrites y1 after it's consumed)
    add_f32<<<ewGrid, 256, 0, stream>>>(feats, ybuf, xout);

    // residual block 2 (x = xout)
    bn_stage(xout, 2);
    conv_stage(2);
    bn_stage(ybuf, 3);
    conv_stage(3);
    add_f32<<<ewGrid, 256, 0, stream>>>(xout, ybuf, xout);
}
